// BidirectionalLSTM_29583734734906
// MI455X (gfx1250) — matrix operand order repaired and checked by reading, not compiled, not run on hardware
//
#include <hip/hip_runtime.h>
#include <stdint.h>

// ---------------------------------------------------------------------------
// Bidirectional LSTM on MI455X (gfx1250).
// All GEMMs on v_wmma_f32_16x16x32_f16; recurrence as a single persistent
// kernel (32 WGs, both directions concurrent, atomic grid barrier per step).
// T=128, B=128, IN=512, H=512, OUT=512, 4H=2048
// ---------------------------------------------------------------------------

typedef __attribute__((ext_vector_type(16))) _Float16 v16h;
typedef __attribute__((ext_vector_type(8)))  _Float16 v8h;
typedef __attribute__((ext_vector_type(4)))  _Float16 v4h;
typedef __attribute__((ext_vector_type(8)))  float    v8f;
typedef __attribute__((ext_vector_type(4)))  float    v4f;

#define TT   128
#define BB   128
#define INS  512
#define HH   512
#define G4   2048   // 4*H
#define OUTS 512
#define KC   1024   // 2*H (concat width)

union Frag16 { v16h v; v8h h[2]; };
union H16x2  { v8h v; _Float16 e[8]; };

__device__ __forceinline__ float sigm_f(float x) { return 1.0f / (1.0f + __expf(-x)); }
__device__ __forceinline__ float tanh_f(float x) { return 2.0f * sigm_f(2.0f * x) - 1.0f; }

// ---------------------------------------------------------------------------
// Convert fp32 weight W[N][K] (row-major) into f16 WMMA-B fragment-major
// layout for B = W^T (K x N). Tile ti = kt*(N/16)+nt holds 512 halfs,
// lane-major: lane l -> col n = nt*16+(l&15), K = kt*32 + (l>>4)*16 + e.
// ---------------------------------------------------------------------------
__global__ __launch_bounds__(256)
void wsw_convert_swizzle_B(const float* __restrict__ W, _Float16* __restrict__ out,
                           int K, int N) {
    int tid  = blockIdx.x * blockDim.x + threadIdx.x;
    int lane = tid & 31;
    int ti   = tid >> 5;
    int ntiles = (K >> 5) * (N >> 4);
    if (ti >= ntiles) return;
    int kt = ti / (N >> 4);
    int nt = ti % (N >> 4);
    int n  = nt * 16 + (lane & 15);
    int k0 = kt * 32 + ((lane >> 4) * 16);
    const float* src = W + (size_t)n * K + k0;
    H16x2 lo, hi;
#pragma unroll
    for (int e = 0; e < 8; ++e) lo.e[e] = (_Float16)src[e];
#pragma unroll
    for (int e = 0; e < 8; ++e) hi.e[e] = (_Float16)src[8 + e];
    v8h* dst = (v8h*)(out + (size_t)ti * 512 + lane * 16);
    dst[0] = lo.v;
    dst[1] = hi.v;
}

// fp32 -> f16 elementwise (n multiple of 4)
__global__ __launch_bounds__(256)
void f32_to_f16_kernel(const float* __restrict__ src, _Float16* __restrict__ dst, int n) {
    int i = (blockIdx.x * blockDim.x + threadIdx.x) * 4;
    if (i >= n) return;
    v4f v = *(const v4f*)(src + i);
    v4h h;
    h.x = (_Float16)v.x; h.y = (_Float16)v.y; h.z = (_Float16)v.z; h.w = (_Float16)v.w;
    *(v4h*)(dst + i) = h;
}

__global__ __launch_bounds__(256)
void zero_u32_kernel(uint32_t* __restrict__ p, int n) {
    int i = blockIdx.x * blockDim.x + threadIdx.x;
    if (i < n) p[i] = 0u;
}

// ---------------------------------------------------------------------------
// WMMA GEMM: C = A16[M][K] * Bsw(KxN, B-fragment swizzled) + bias.
// 256 threads = 8 waves; WG tile 64(M) x 128(N); wave -> (wm = 16-row
// subtile, wn = 64-col half, 4 N-tiles each).
// rev:      remap A row m -> (T-1 - m/B)*B + m%B (time-reversed input).
// fragout:  1 -> store C-fragment-major (per-lane v8f, tile = (m/16)*(N/16)
//           + n/16, 256 floats/tile); 0 -> row-major fp32.
// ---------------------------------------------------------------------------
__global__ __launch_bounds__(256)
void gemm_bias_wmma(const _Float16* __restrict__ A, const _Float16* __restrict__ Bsw,
                    const float* __restrict__ bias0, const float* __restrict__ bias1,
                    float* __restrict__ C, int M, int N, int K, int rev, int fragout) {
    __shared__ __align__(16) _Float16 As[64 * 40];   // 64 rows x 32 K, pad to 40 halfs

    const int tid  = threadIdx.x;
    const int lane = tid & 31;
    const int wave = tid >> 5;
    const int wm   = wave >> 1;
    const int wn   = wave & 1;
    const int lo   = lane & 15;
    const int hi   = lane >> 4;

    const int mblk = blockIdx.y * 64;
    const int nblk = blockIdx.x * 128 + wn * 64;

    v8f acc[4];
#pragma unroll
    for (int j = 0; j < 4; ++j) {
        int n = nblk + j * 16 + lo;
        float b = bias0[n];
        if (bias1) b += bias1[n];
#pragma unroll
        for (int v = 0; v < 8; ++v) acc[j][v] = b;
    }

    const int srow = tid >> 2;       // 0..63
    const int sch  = tid & 3;        // 16B chunk
    int gm = mblk + srow;
    int arow = gm;
    if (rev) arow = (TT - 1 - gm / BB) * BB + (gm % BB);
    const _Float16* arp = A + (size_t)arow * K + sch * 8;
    _Float16* lwp = &As[srow * 40 + sch * 8];

    const int ksteps = K >> 5;
    for (int kk = 0; kk < ksteps; ++kk) {
        __syncthreads();
        *(v8h*)lwp = *(const v8h*)(arp + kk * 32);
        __syncthreads();

        Frag16 af;
        const _Float16* ap = &As[(wm * 16 + lo) * 40 + hi * 16];
        af.h[0] = *(const v8h*)ap;
        af.h[1] = *(const v8h*)(ap + 8);

#pragma unroll
        for (int j = 0; j < 4; ++j) {
            int nt = (nblk + j * 16) >> 4;
            const _Float16* bp = Bsw + ((size_t)kk * (N >> 4) + nt) * 512 + lane * 16;
            Frag16 bf;
            bf.h[0] = *(const v8h*)bp;
            bf.h[1] = *(const v8h*)(bp + 8);
            acc[j] = __builtin_amdgcn_wmma_f32_16x16x32_f16(
                false, af.v, false, bf.v, (short)0, acc[j], false, false);
        }
    }

    if (fragout) {
        int MT = blockIdx.y * 4 + wm;            // global 16-row tile index
#pragma unroll
        for (int j = 0; j < 4; ++j) {
            int nt = (nblk + j * 16) >> 4;
            float* dst = C + ((size_t)MT * (N >> 4) + nt) * 256 + lane * 8;
            *(v8f*)dst = acc[j];
        }
    } else {
#pragma unroll
        for (int j = 0; j < 4; ++j) {
            int n = nblk + j * 16 + lo;
#pragma unroll
            for (int v = 0; v < 8; ++v) {
                int m = mblk + wm * 16 + v + hi * 8;
                C[(size_t)m * N + n] = acc[j][v];
            }
        }
    }
}

// ---------------------------------------------------------------------------
// Persistent bidirectional recurrence. Grid (8, 2, 2) = 32 WGs:
// blockIdx.x = hidden-column group (64), .y = batch-row group (64), .z = dir.
// Each WG owns matching i/f/g/o columns (n, n+512, n+1024, n+1536), loops all
// 128 timesteps; steps synchronized with a device-scope atomic barrier.
// xw and c are C-fragment-major (per-lane contiguous v8f).
// ---------------------------------------------------------------------------
__global__ __launch_bounds__(256)
void lstm_recur_persistent(const _Float16* __restrict__ BswHhF,
                           const _Float16* __restrict__ BswHhB,
                           const float* __restrict__ xwF,
                           const float* __restrict__ xwB,
                           float* __restrict__ cF, float* __restrict__ cB,
                           _Float16* __restrict__ hF0, _Float16* __restrict__ hF1,
                           _Float16* __restrict__ hB0, _Float16* __restrict__ hB1,
                           _Float16* __restrict__ hcat,
                           unsigned* __restrict__ bar) {
    __shared__ __align__(16) _Float16 As[64 * 40];

    const int tid  = threadIdx.x;
    const int lane = tid & 31;
    const int wave = tid >> 5;
    const int wm   = wave >> 1;
    const int wn   = wave & 1;
    const int lo   = lane & 15;
    const int hi   = lane >> 4;
    const int dir  = blockIdx.z;

    const _Float16* Bsw = dir ? BswHhB : BswHhF;
    const float*    xw  = dir ? xwB : xwF;
    float*          cst = dir ? cB : cF;
    _Float16* hpp[2];
    hpp[0] = dir ? hB0 : hF0;
    hpp[1] = dir ? hB1 : hF1;

    const int mblk = blockIdx.y * 64;
    const int mb   = mblk + wm * 16;
    const int mt   = blockIdx.y * 4 + wm;         // 16-row tile index (0..7)
    const int cb0  = blockIdx.x * 64 + wn * 32;   // hidden-index base

    const int srow = tid >> 2;
    const int sch  = tid & 3;
    _Float16* lwp = &As[srow * 40 + sch * 8];

    for (int t = 0; t < TT; ++t) {
        const _Float16* hin = hpp[t & 1];
        _Float16* hout      = hpp[(t + 1) & 1];
        const float* xw_t   = xw + (size_t)t * BB * G4;   // frag-major, same extent

        // Seed accumulators from xw_t (fragment-major: contiguous per lane)
        v8f acc[4][2];
#pragma unroll
        for (int g = 0; g < 4; ++g)
#pragma unroll
            for (int j = 0; j < 2; ++j) {
                int nt = (g * HH + cb0 + j * 16) >> 4;
                acc[g][j] = *(const v8f*)(xw_t + ((size_t)mt * (G4 >> 4) + nt) * 256 + lane * 8);
            }

        const _Float16* arp = hin + (size_t)(mblk + srow) * HH + sch * 8;

        for (int kk = 0; kk < (HH >> 5); ++kk) {
            __syncthreads();
            *(v8h*)lwp = *(const v8h*)(arp + kk * 32);
            __syncthreads();

            Frag16 af;
            const _Float16* ap = &As[(wm * 16 + lo) * 40 + hi * 16];
            af.h[0] = *(const v8h*)ap;
            af.h[1] = *(const v8h*)(ap + 8);

#pragma unroll
            for (int g = 0; g < 4; ++g)
#pragma unroll
                for (int j = 0; j < 2; ++j) {
                    int nt = (g * HH + cb0 + j * 16) >> 4;
                    const _Float16* bp = Bsw + ((size_t)kk * (G4 >> 4) + nt) * 512 + lane * 16;
                    Frag16 bf;
                    bf.h[0] = *(const v8h*)bp;
                    bf.h[1] = *(const v8h*)(bp + 8);
                    acc[g][j] = __builtin_amdgcn_wmma_f32_16x16x32_f16(
                        false, af.v, false, bf.v, (short)0, acc[g][j], false, false);
                }
        }

        // Gate nonlinearities + state update (c fragment-major: v8f RMW)
        int tout = dir ? (TT - 1 - t) : t;
        _Float16* hcat_t = hcat + (size_t)tout * BB * KC + dir * HH;
#pragma unroll
        for (int j = 0; j < 2; ++j) {
            int ct = (cb0 + j * 16) >> 4;         // hidden 16-col tile (0..31)
            float* cp = cst + ((size_t)mt * (HH >> 4) + ct) * 256 + lane * 8;
            v8f cold = *(v8f*)cp;
            v8f cnew;
            int hidx = cb0 + j * 16 + lo;
#pragma unroll
            for (int v = 0; v < 8; ++v) {
                float ig = sigm_f(acc[0][j][v]);
                float fg = sigm_f(acc[1][j][v]);
                float gg = tanh_f(acc[2][j][v]);
                float og = sigm_f(acc[3][j][v]);
                float cv = fg * cold[v] + ig * gg;
                cnew[v] = cv;
                float h = og * tanh_f(cv);
                int m = mb + v + hi * 8;
                hout[(size_t)m * HH + hidx]   = (_Float16)h;
                hcat_t[(size_t)m * KC + hidx] = (_Float16)h;
            }
            *(v8f*)cp = cnew;
        }

        // Device-wide step barrier (32 WGs): publish h, then wait for all.
        __threadfence();
        __syncthreads();
        if (tid == 0) {
            atomicAdd(bar, 1u);
            unsigned target = 32u * (unsigned)(t + 1);
            while (atomicAdd(bar, 0u) < target) __builtin_amdgcn_s_sleep(2);
        }
        __syncthreads();
        __threadfence();   // invalidate WGP$ so next step sees fresh h
    }
}

// ---------------------------------------------------------------------------
// Host-side orchestration (11 launches total)
// ---------------------------------------------------------------------------
extern "C" void kernel_launch(void* const* d_in, const int* in_sizes, int n_in,
                              void* d_out, int out_size, void* d_ws, size_t ws_size,
                              hipStream_t stream) {
    const float* x      = (const float*)d_in[0];
    const float* W_ih_f = (const float*)d_in[1];
    const float* W_hh_f = (const float*)d_in[2];
    const float* b_ih_f = (const float*)d_in[3];
    const float* b_hh_f = (const float*)d_in[4];
    const float* W_ih_b = (const float*)d_in[5];
    const float* W_hh_b = (const float*)d_in[6];
    const float* b_ih_b = (const float*)d_in[7];
    const float* b_hh_b = (const float*)d_in[8];
    const float* W_lin  = (const float*)d_in[9];
    const float* b_lin  = (const float*)d_in[10];
    float* out = (float*)d_out;

    size_t off = 0;
    auto carve = [&](size_t bytes) -> char* {
        char* p = (char*)d_ws + off;
        off += (bytes + 255) & ~(size_t)255;
        return p;
    };
    _Float16* BswIhF = (_Float16*)carve((size_t)INS * G4 * 2);
    _Float16* BswIhB = (_Float16*)carve((size_t)INS * G4 * 2);
    _Float16* BswHhF = (_Float16*)carve((size_t)HH * G4 * 2);
    _Float16* BswHhB = (_Float16*)carve((size_t)HH * G4 * 2);
    _Float16* BswLin = (_Float16*)carve((size_t)KC * OUTS * 2);
    _Float16* x16    = (_Float16*)carve((size_t)TT * BB * INS * 2);
    float*    xwF    = (float*)carve((size_t)TT * BB * G4 * 4);   // frag-major
    float*    xwB    = (float*)carve((size_t)TT * BB * G4 * 4);   // frag-major
    _Float16* hSt    = (_Float16*)carve((size_t)4 * BB * HH * 2); // [dir][pp]
    float*    cSt    = (float*)carve((size_t)2 * BB * HH * 4);    // frag-major
    unsigned* bar    = (unsigned*)carve(256);
    _Float16* hcat   = (_Float16*)carve((size_t)TT * BB * KC * 2);

    // 1) Swizzle weights into WMMA B-fragment layout
    {
        int thr, blk;
        thr = INS * G4 / 16; blk = (thr + 255) / 256;
        wsw_convert_swizzle_B<<<blk, 256, 0, stream>>>(W_ih_f, BswIhF, INS, G4);
        wsw_convert_swizzle_B<<<blk, 256, 0, stream>>>(W_ih_b, BswIhB, INS, G4);
        wsw_convert_swizzle_B<<<blk, 256, 0, stream>>>(W_hh_f, BswHhF, HH, G4);
        wsw_convert_swizzle_B<<<blk, 256, 0, stream>>>(W_hh_b, BswHhB, HH, G4);
        thr = KC * OUTS / 16; blk = (thr + 255) / 256;
        wsw_convert_swizzle_B<<<blk, 256, 0, stream>>>(W_lin, BswLin, KC, OUTS);
    }

    // 2) x -> f16
    {
        int n = TT * BB * INS;
        f32_to_f16_kernel<<<(n / 4 + 255) / 256, 256, 0, stream>>>(x, x16, n);
    }

    // 3) Zero h ping-pong, c state, and barrier counter (contiguous carve)
    {
        int ndw = (int)(((size_t)4 * BB * HH * 2 + (size_t)2 * BB * HH * 4 + 256) / 4);
        zero_u32_kernel<<<(ndw + 255) / 256, 256, 0, stream>>>((uint32_t*)hSt, ndw);
    }

    // 4) xw = x * W_ih^T + b_ih + b_hh, both directions, fragment-major out
    {
        dim3 grid(G4 / 128, (TT * BB) / 64);
        gemm_bias_wmma<<<grid, 256, 0, stream>>>(x16, BswIhF, b_ih_f, b_hh_f,
                                                 xwF, TT * BB, G4, INS, 0, 1);
        gemm_bias_wmma<<<grid, 256, 0, stream>>>(x16, BswIhB, b_ih_b, b_hh_b,
                                                 xwB, TT * BB, G4, INS, 1, 1);
    }

    // 5) Persistent recurrence: both directions, all 128 steps, one launch
    {
        dim3 grid(HH / 64, BB / 64, 2);   // 32 workgroups
        lstm_recur_persistent<<<grid, 256, 0, stream>>>(
            BswHhF, BswHhB, xwF, xwB,
            cSt, cSt + BB * HH,
            hSt, hSt + BB * HH, hSt + 2 * BB * HH, hSt + 3 * BB * HH,
            hcat, bar);
    }

    // 6) out = [h_f | h_b] * W_lin^T + b_lin (row-major fp32 out)
    {
        dim3 grid(OUTS / 128, (TT * BB) / 64);
        gemm_bias_wmma<<<grid, 256, 0, stream>>>(hcat, BswLin, b_lin, nullptr,
                                                 out, TT * BB, OUTS, KC, 0, 0);
    }
    (void)in_sizes; (void)n_in; (void)out_size; (void)ws_size;
}
